// GCNEncoder_59193239273691
// MI455X (gfx1250) — matrix operand order repaired and checked by reading, not compiled, not run on hardware
//
#include <hip/hip_runtime.h>
#include <cstdint>
#include <cstddef>

// CDNA5 wave32 WMMA vector types
typedef __attribute__((ext_vector_type(16))) __bf16 v16bf;
typedef __attribute__((ext_vector_type(8)))  float  v8f;

// ---------------------------------------------------------------------------
// utility kernels
// ---------------------------------------------------------------------------
__global__ void fill_kernel(float* __restrict__ p, float v, size_t n) {
  size_t i = (size_t)blockIdx.x * blockDim.x + threadIdx.x;
  if (i < n) p[i] = v;
}

// deg[dst[e]] += 1  (deg pre-filled with 1.0 for the self-loop)
__global__ void degree_kernel(const int* __restrict__ dst,
                              float* __restrict__ deg, int nE) {
  int e = blockIdx.x * blockDim.x + threadIdx.x;
  if (e < nE) unsafeAtomicAdd(&deg[dst[e]], 1.0f);
}

__global__ void rsqrt_kernel(const float* __restrict__ deg,
                             float* __restrict__ dinv, int n) {
  int i = blockIdx.x * blockDim.x + threadIdx.x;
  if (i < n) {
    float d = deg[i];
    dinv[i] = (d > 0.0f) ? rsqrtf(d) : 0.0f;
  }
}

// norm[e] = dinv[src[e]] * dinv[dst[e]]  (computed once, used by both layers)
__global__ void norm_kernel(const int* __restrict__ src,
                            const int* __restrict__ dst,
                            const float* __restrict__ dinv,
                            float* __restrict__ norm, int nE) {
  int e = blockIdx.x * blockDim.x + threadIdx.x;
  if (e < nE) norm[e] = dinv[src[e]] * dinv[dst[e]];
}

// ---------------------------------------------------------------------------
// Pre-swizzle W[K,N] (f32 row-major) into WMMA B-fragment layout, bf16 hi/lo:
//   frag layout: [ts = kstep*(N/16) + ntile][lane 0..31][j 0..15]
//   element (lane,j) = W[kstep*32 + (j&7) + 8*(lane>>4) + 16*(j>>3)][ntile*16 + (lane&15)]
// ---------------------------------------------------------------------------
__global__ void prep_w_kernel(const float* __restrict__ W,
                              unsigned short* __restrict__ Wh_u,
                              unsigned short* __restrict__ Wl_u,
                              int K, int N) {
  __bf16* __restrict__ Wh = (__bf16*)Wh_u;
  __bf16* __restrict__ Wl = (__bf16*)Wl_u;
  int idx = blockIdx.x * blockDim.x + threadIdx.x;
  int total = K * N;                       // == (K/32)*(N/16)*32*16
  if (idx >= total) return;
  int j    = idx & 15;
  int lane = (idx >> 4) & 31;
  int ts   = idx >> 9;
  int nT16 = N >> 4;
  int nt   = ts % nT16;
  int ks   = ts / nT16;
  int lo = lane & 15, hi = lane >> 4;
  int k = (ks << 5) + (j & 7) + (hi << 3) + ((j >> 3) << 4);
  int n = (nt << 4) + lo;
  float v = W[(size_t)k * N + n];
  __bf16 hv = (__bf16)v;                   // RNE hi part
  Wh[idx] = hv;
  Wl[idx] = (__bf16)(v - (float)hv);       // residual lo part
}

// ---------------------------------------------------------------------------
// GEMM  D[M,N] = A[M,K] @ W[K,N], W pre-swizzled bf16 hi/lo fragments.
// bf16x3 split (f32 accumulate): D ~= ah*bh + ah*bl + al*bh.
// One wave computes a 16x64 tile (4 x v8f accumulators).
//  - A: 4 x b128 per k-step via a running pointer + immediate offsets.
//  - B: 16 x b128 per k-step; the 4 N-tiles are contiguous (stride 1024B),
//       so a single running pointer per array + immediate offsets suffices.
//  - WMMAs grouped term-major so consecutive WMMAs target different
//    accumulators (no same-D back-to-back => no hazard NOPs).
// Requires M%16==0, K%32==0, N%64==0.
// ---------------------------------------------------------------------------
__global__ void __launch_bounds__(128)
gemm_bf16x3_wmma(const float* __restrict__ A,
                 const unsigned short* __restrict__ Bh_u,
                 const unsigned short* __restrict__ Bl_u,
                 float* __restrict__ D, int M, int K, int N) {
  const int lane = threadIdx.x & 31;
  const int wid  = blockIdx.x * (blockDim.x >> 5) + (threadIdx.x >> 5);
  const int nTilesN = N >> 6;                  // 64-wide column tiles
  const int tileM = wid / nTilesN;
  const int tileN = wid - tileM * nTilesN;
  if (tileM * 16 >= M) return;
  const int m0 = tileM << 4;
  const int n0 = tileN << 6;
  const int hi = lane >> 4;
  const int lo = lane & 15;
  const int nT16 = N >> 4;

  v8f acc[4] = {};

  // running pointers (advance by constant strides; loads use imm offsets)
  const float4* __restrict__ ap =
      (const float4*)(A + (size_t)(m0 + lo) * K) + (hi << 1);
  const size_t lane_frag = ((size_t)(n0 >> 4) * 32 + lane) * 16;
  const __bf16* __restrict__ bhp = (const __bf16*)Bh_u + lane_frag;
  const __bf16* __restrict__ blp = (const __bf16*)Bl_u + lane_frag;
  const size_t kstride = (size_t)nT16 * 512;   // bf16 elems per k-step

  const int nKsteps = K >> 5;
  for (int ks = 0; ks < nKsteps; ++ks) {
    // ---- A fragment: lanes 0-15 need k0+{0..7,16..23}; lanes 16-31 +8.
    float4 f0 = ap[0];
    float4 f1 = ap[1];
    float4 f2 = ap[4];
    float4 f3 = ap[5];
    ap += 8;                                   // 32 floats per k-step
    // ---- B fragments: 4 tiles x (hi,lo), contiguous at t*512 elements
    v16bf bhf[4], blf[4];
#pragma unroll
    for (int t = 0; t < 4; ++t) {
      bhf[t] = *(const v16bf*)(bhp + t * 512);
      blf[t] = *(const v16bf*)(blp + t * 512);
    }
    bhp += kstride;
    blp += kstride;

    float af[16] = {f0.x, f0.y, f0.z, f0.w, f1.x, f1.y, f1.z, f1.w,
                    f2.x, f2.y, f2.z, f2.w, f3.x, f3.y, f3.z, f3.w};
    v16bf ah, al;
#pragma unroll
    for (int j = 0; j < 16; ++j) {
      const __bf16 hv = (__bf16)af[j];
      ah[j] = hv;
      al[j] = (__bf16)(af[j] - (float)hv);
    }

    // term-major WMMA issue: 4 independent D's between same-D reuses
#pragma unroll
    for (int t = 0; t < 4; ++t)
      acc[t] = __builtin_amdgcn_wmma_f32_16x16x32_bf16(
          false, ah, false, bhf[t], (short)0, acc[t], false, false);
#pragma unroll
    for (int t = 0; t < 4; ++t)
      acc[t] = __builtin_amdgcn_wmma_f32_16x16x32_bf16(
          false, ah, false, blf[t], (short)0, acc[t], false, false);
#pragma unroll
    for (int t = 0; t < 4; ++t)
      acc[t] = __builtin_amdgcn_wmma_f32_16x16x32_bf16(
          false, al, false, bhf[t], (short)0, acc[t], false, false);
  }

  // ---- D store: vgpr r of v8f -> row = r + 8*hi, col = lo (per 16x16 tile)
#pragma unroll
  for (int t = 0; t < 4; ++t)
#pragma unroll
    for (int r = 0; r < 8; ++r)
      D[(size_t)(m0 + r + (hi << 3)) * N + (n0 + (t << 4) + lo)] = acc[t][r];
}

// ---------------------------------------------------------------------------
// Edge aggregation: acc[dst] += h[src] * norm[e]
// One wave per edge; float4 gathers (coalesced b128), scalar f32 atomics.
// Requires C % 128 == 0 (256 and 128 both OK).
// ---------------------------------------------------------------------------
__global__ void aggregate_kernel(const float* __restrict__ h,
                                 const int* __restrict__ src,
                                 const int* __restrict__ dst,
                                 const float* __restrict__ norm,
                                 float* __restrict__ acc, int nE, int C) {
  const int wid  = (blockIdx.x * blockDim.x + threadIdx.x) >> 5;
  const int lane = threadIdx.x & 31;
  if (wid >= nE) return;
  const int s = src[wid];
  const int d = dst[wid];
  const float nm = norm[wid];
  const float4* __restrict__ hs = (const float4*)(h + (size_t)s * C);
  float* __restrict__ ad = acc + (size_t)d * C;
  const int nq = C >> 7;                       // float4 chunks per lane
  for (int q = 0; q < nq; ++q) {
    const int c4 = lane + (q << 5);
    const float4 v = hs[c4];
    const int c = c4 << 2;
    unsafeAtomicAdd(&ad[c + 0], v.x * nm);
    unsafeAtomicAdd(&ad[c + 1], v.y * nm);
    unsafeAtomicAdd(&ad[c + 2], v.z * nm);
    unsafeAtomicAdd(&ad[c + 3], v.w * nm);
  }
}

// ---------------------------------------------------------------------------
// Finalize: acc = [relu]( acc + h*dinv[node]^2 + bias[c] )  (self-loop+bias)
// ---------------------------------------------------------------------------
__global__ void finalize_kernel(float* __restrict__ acc,
                                const float* __restrict__ h,
                                const float* __restrict__ dinv,
                                const float* __restrict__ bias,
                                int C, size_t total4, int do_relu) {
  size_t i = (size_t)blockIdx.x * blockDim.x + threadIdx.x;
  if (i >= total4) return;
  const int C4 = C >> 2;
  size_t node = i / (size_t)C4;
  int c4 = (int)(i - node * (size_t)C4);
  float dn = dinv[node];
  float sn = dn * dn;
  float4 a = ((const float4*)acc)[i];
  float4 hh = ((const float4*)h)[i];
  float4 b = ((const float4*)bias)[c4];
  float4 r;
  r.x = a.x + hh.x * sn + b.x;
  r.y = a.y + hh.y * sn + b.y;
  r.z = a.z + hh.z * sn + b.z;
  r.w = a.w + hh.w * sn + b.w;
  if (do_relu) {
    r.x = fmaxf(r.x, 0.0f); r.y = fmaxf(r.y, 0.0f);
    r.z = fmaxf(r.z, 0.0f); r.w = fmaxf(r.w, 0.0f);
  }
  ((float4*)acc)[i] = r;
}

// ---------------------------------------------------------------------------
// host side
// ---------------------------------------------------------------------------
static inline unsigned nblk(size_t threads, int bs) {
  return (unsigned)((threads + (size_t)bs - 1) / (size_t)bs);
}

extern "C" void kernel_launch(void* const* d_in, const int* in_sizes, int n_in,
                              void* d_out, int out_size, void* d_ws, size_t ws_size,
                              hipStream_t stream) {
  const float* x  = (const float*)d_in[0];
  const int*   ei = (const int*)d_in[1];   // [2,E]: row 0 = src, row 1 = dst
  const float* W1 = (const float*)d_in[2];
  const float* b1 = (const float*)d_in[3];
  const float* W2 = (const float*)d_in[4];
  const float* b2 = (const float*)d_in[5];

  const int hidden = in_sizes[3];               // 256
  const int outc   = in_sizes[5];               // 128
  const int inc    = in_sizes[2] / hidden;      // 512
  const int n      = in_sizes[0] / inc;         // 50000
  const int E      = in_sizes[1] / 2;           // 800000
  const int* src = ei;
  const int* dst = ei + E;

  // workspace layout (~132 MB); all sub-buffers stay 32B-aligned.
  float* deg  = (float*)d_ws;                   // [n]
  float* dinv = deg + n;                        // [n]
  float* h    = dinv + n;                       // [n, hidden]   x@W1
  float* acc1 = h + (size_t)n * hidden;         // [n, hidden]   layer-1 out
  float* h2   = acc1 + (size_t)n * hidden;      // [n, outc]     h1@W2
  float* norm = h2 + (size_t)n * outc;          // [E]
  unsigned short* Wh1 = (unsigned short*)(norm + E);
  unsigned short* Wl1 = Wh1 + (size_t)inc * hidden;
  unsigned short* Wh2 = Wl1 + (size_t)inc * hidden;
  unsigned short* Wl2 = Wh2 + (size_t)hidden * outc;
  float* out  = (float*)d_out;                  // [n, outc]

  // init: deg=1 (self-loop), accumulators zeroed; weight pre-swizzle
  fill_kernel<<<nblk(n, 256), 256, 0, stream>>>(deg, 1.0f, (size_t)n);
  fill_kernel<<<nblk((size_t)n * hidden, 256), 256, 0, stream>>>(
      acc1, 0.0f, (size_t)n * hidden);
  fill_kernel<<<nblk((size_t)n * outc, 256), 256, 0, stream>>>(
      out, 0.0f, (size_t)n * outc);
  prep_w_kernel<<<nblk((size_t)inc * hidden, 256), 256, 0, stream>>>(
      W1, Wh1, Wl1, inc, hidden);
  prep_w_kernel<<<nblk((size_t)hidden * outc, 256), 256, 0, stream>>>(
      W2, Wh2, Wl2, hidden, outc);

  // degree -> dinv -> per-edge norm
  degree_kernel<<<nblk(E, 256), 256, 0, stream>>>(dst, deg, E);
  rsqrt_kernel<<<nblk(n, 256), 256, 0, stream>>>(deg, dinv, n);
  norm_kernel<<<nblk(E, 256), 256, 0, stream>>>(src, dst, dinv, norm, E);

  // layer 1: h = x @ W1 (WMMA bf16x3), aggregate, finalize(+b1, relu)
  {
    size_t waves = (size_t)(n / 16) * (size_t)(hidden / 64);
    gemm_bf16x3_wmma<<<nblk(waves * 32, 128), 128, 0, stream>>>(
        x, Wh1, Wl1, h, n, inc, hidden);
  }
  aggregate_kernel<<<nblk((size_t)E * 32, 256), 256, 0, stream>>>(
      h, src, dst, norm, acc1, E, hidden);
  finalize_kernel<<<nblk((size_t)n * hidden / 4, 256), 256, 0, stream>>>(
      acc1, h, dinv, b1, hidden, (size_t)n * hidden / 4, 1);

  // layer 2: h2 = h1 @ W2, aggregate into d_out, finalize(+b2, no relu)
  {
    size_t waves = (size_t)(n / 16) * (size_t)(outc / 64);
    gemm_bf16x3_wmma<<<nblk(waves * 32, 128), 128, 0, stream>>>(
        acc1, Wh2, Wl2, h2, n, hidden, outc);
  }
  aggregate_kernel<<<nblk((size_t)E * 32, 256), 256, 0, stream>>>(
      h2, src, dst, norm, out, E, outc);
  finalize_kernel<<<nblk((size_t)n * outc / 4, 256), 256, 0, stream>>>(
      out, h2, dinv, b2, outc, (size_t)n * outc / 4, 0);
}
